// MMoE_48902497632411
// MI455X (gfx1250) — matrix operand order repaired and checked by reading, not compiled, not run on hardware
//
#include <hip/hip_runtime.h>
#include <math.h>

#define TT 3
#define BB 4096
#define IN_DIM 1024
#define HID_DIM 2048
#define OUT_DIM 1024
#define NE 16
#define TOPK 4
#define CHUNK 1024
#define EPSV 2.220446049250313e-16f
#define LSTR 20   // LDS row stride in u32 (16 data pairs + pad, 16B-aligned rows)

typedef __attribute__((ext_vector_type(16))) __bf16 v16bf;
typedef __attribute__((ext_vector_type(8))) float v8f;

union FragBF { v16bf v; unsigned u[8]; uint4 q[2]; };
union Acc8 { v8f v; float f[8]; };

// fp32 -> packed bf16 pair. HW packed convert if present, else round + v_perm.
#if __has_builtin(__builtin_amdgcn_cvt_pk_bf16_f32)
typedef __attribute__((ext_vector_type(2))) __bf16 v2bf;
__device__ __forceinline__ unsigned pack_bf16(float a, float b) {
  union { v2bf v; unsigned u; } c;
  c.v = __builtin_amdgcn_cvt_pk_bf16_f32(a, b);
  return c.u;
}
#else
__device__ __forceinline__ unsigned pack_bf16(float a, float b) {
  unsigned ua = __float_as_uint(a) + 0x8000u;   // round-half-up to bf16
  unsigned ub = __float_as_uint(b) + 0x8000u;
  return __builtin_amdgcn_perm(ub, ua, 0x07060302u);  // {ub[31:16], ua[31:16]}
}
#endif

// 16-bit A-fragment (16x32), k order used here: u[0..7] = 16 contiguous K at K=half*16
__device__ __forceinline__ v16bf load_frag_A(const unsigned* lds, int row, int half) {
  FragBF f;
  const uint4* p = (const uint4*)(lds + row * LSTR + half * 8);
  f.q[0] = p[0];        // pairs half*8 + 0..3
  f.q[1] = p[1];        // pairs half*8 + 4..7
  return f.v;
}

// 16-bit B-fragment (32x16), k order used here: u[0..7] = 16 contiguous K at K=half*16
__device__ __forceinline__ v16bf load_frag_B(const unsigned* lds, int row, int half) {
  FragBF f;
  const uint4* p = (const uint4*)(lds + row * LSTR + half * 8);
  f.q[0] = p[0];
  f.q[1] = p[1];
  return f.v;
}

// B-fragment straight from row-major global bf16 [n][k]: two b128 loads.
__device__ __forceinline__ v16bf load_frag_B_global(const unsigned short* rowPtr, int kk, int half) {
  FragBF f;
  const uint4* p = (const uint4*)(rowPtr + kk + half * 16);
  f.q[0] = p[0];
  f.q[1] = p[1];
  return f.v;
}

// ---------------- one-time fp32 -> bf16 pre-conversion ------------------
__global__ __launch_bounds__(256) void cvt_bf16_kernel(
    const float* __restrict__ src, unsigned* __restrict__ dst, int nQuads) {
  int stride = gridDim.x * 256;
  for (int i = blockIdx.x * 256 + threadIdx.x; i < nQuads; i += stride) {
    float4 v = *(const float4*)(src + (size_t)i * 4);
    *(uint2*)(dst + (size_t)i * 2) =
        make_uint2(pack_bf16(v.x, v.y), pack_bf16(v.z, v.w));
  }
}

// ---------------- gating: logits = einsum('bi,tie->tbe') ----------------
__global__ __launch_bounds__(256) void gating_logits_kernel(
    const float* __restrict__ x, const float* __restrict__ wg,
    float* __restrict__ logits) {
  int gid = blockIdx.x * 256 + threadIdx.x;
  int e = gid & (NE - 1);
  int b = (gid >> 4) & (BB - 1);
  int t = gid >> 16;
  if (t >= TT) return;
  const float* xr = x + (size_t)b * IN_DIM;
  const float* w = wg + (size_t)t * IN_DIM * NE + e;
  float s = 0.f;
#pragma unroll 8
  for (int i = 0; i < IN_DIM; ++i) s = fmaf(xr[i], w[(size_t)i * NE], s);
  logits[((size_t)t * BB + b) * NE + e] = s;
}

// ---------------- gating: top-4 softmax, threshold, dense scatter -------
__global__ __launch_bounds__(256) void gating_topk_kernel(
    const float* __restrict__ logits, float* __restrict__ gates) {
  int b = blockIdx.x * 256 + threadIdx.x;
  if (b >= BB) return;
  float g[NE][TT];
#pragma unroll
  for (int e = 0; e < NE; ++e)
#pragma unroll
    for (int t = 0; t < TT; ++t) g[e][t] = 0.f;

  for (int t = 0; t < TT; ++t) {
    float l[NE];
#pragma unroll
    for (int e = 0; e < NE; ++e) l[e] = logits[((size_t)t * BB + b) * NE + e];
    int ti[TOPK]; float tv[TOPK];
#pragma unroll
    for (int k = 0; k < TOPK; ++k) {
      float best = -3.4e38f; int bi = 0;
#pragma unroll
      for (int e = 0; e < NE; ++e) {
        bool used = false;
#pragma unroll
        for (int j = 0; j < TOPK; ++j) used = used || ((j < k) && (ti[j] == e));
        if (!used && l[e] > best) { best = l[e]; bi = e; }
      }
      tv[k] = best; ti[k] = bi;
    }
    float m = tv[0];
    float den = 0.f, ex[TOPK];
#pragma unroll
    for (int k = 0; k < TOPK; ++k) { ex[k] = __expf(tv[k] - m); den += ex[k]; }
#pragma unroll
    for (int k = 0; k < TOPK; ++k) {
      float gv = ex[k] / den;
      if (gv <= 1e-4f) gv = 0.f;
      g[ti[k]][t] = gv;
    }
  }
#pragma unroll
  for (int e = 0; e < NE; ++e)
#pragma unroll
    for (int t = 0; t < TT; ++t)
      gates[((size_t)b * NE + e) * TT + t] = g[e][t];
}

// ------- GEMM1: D[h][b] = W1 * X^T (+b1, relu) -> Ht[e][b][h] bf16 ------
// Block 128(M=h) x 128(N=b), 8 waves (4Mx2N), wave 32x64, K-step 32,
// double-buffered LDS + register prefetch. PRE: sources already bf16-packed.
template <bool PRE>
__global__ __launch_bounds__(256, 2) void gemm1_kernel(
    const void* __restrict__ xsrc, const void* __restrict__ w1src,
    const float* __restrict__ fc1_b, unsigned short* __restrict__ Ht,
    int chunkBase) {
  __shared__ __attribute__((aligned(16))) unsigned ldsA[2][128 * LSTR];
  __shared__ __attribute__((aligned(16))) unsigned ldsB[2][128 * LSTR];

  const int e = blockIdx.z;
  const int nBase = blockIdx.x * 128;
  const int mBase = blockIdx.y * 128;
  const int tid = threadIdx.x;
  const int lane = tid & 31;
  const int wave = tid >> 5;
  const int waveM = wave & 3;
  const int waveN = wave >> 2;
  const int half = lane >> 4;
  const int l16 = lane & 15;

  const float* aF = (const float*)w1src + ((size_t)e * HID_DIM + mBase) * IN_DIM;
  const float* bF = (const float*)xsrc + (size_t)(chunkBase + nBase) * IN_DIM;
  const unsigned* aU = (const unsigned*)w1src + (((size_t)e * HID_DIM + mBase) * IN_DIM >> 1);
  const unsigned* bU = (const unsigned*)xsrc + ((size_t)(chunkBase + nBase) * IN_DIM >> 1);

  v8f acc[2][4];
  const v8f vzero = {0.f, 0.f, 0.f, 0.f, 0.f, 0.f, 0.f, 0.f};
#pragma unroll
  for (int i = 0; i < 2; ++i)
#pragma unroll
    for (int j = 0; j < 4; ++j) acc[i][j] = vzero;

  float4 pa0, pa1, pa2, pa3, pb0, pb1, pb2, pb3;   // fp32 staging
  uint4 qa0, qa1, qb0, qb1;                        // bf16 staging
  const int rowP = tid >> 1;           // PRE path: 0..127
  const int colP = (tid & 1) * 8;      // u32 col: 0 or 8

  auto loadStage = [&](int kk) {
    if constexpr (PRE) {
      const uint4* pA = (const uint4*)(aU + (size_t)rowP * (IN_DIM / 2) + (kk >> 1) + colP);
      qa0 = pA[0]; qa1 = pA[1];
      const uint4* pB = (const uint4*)(bU + (size_t)rowP * (IN_DIM / 2) + (kk >> 1) + colP);
      qb0 = pB[0]; qb1 = pB[1];
    } else {
      int row = tid >> 3, q = tid & 7;
      pa0 = *(const float4*)(aF + (size_t)row * IN_DIM + kk + q * 4);
      pb0 = *(const float4*)(bF + (size_t)row * IN_DIM + kk + q * 4);
      pa1 = *(const float4*)(aF + (size_t)(row + 32) * IN_DIM + kk + q * 4);
      pb1 = *(const float4*)(bF + (size_t)(row + 32) * IN_DIM + kk + q * 4);
      pa2 = *(const float4*)(aF + (size_t)(row + 64) * IN_DIM + kk + q * 4);
      pb2 = *(const float4*)(bF + (size_t)(row + 64) * IN_DIM + kk + q * 4);
      pa3 = *(const float4*)(aF + (size_t)(row + 96) * IN_DIM + kk + q * 4);
      pb3 = *(const float4*)(bF + (size_t)(row + 96) * IN_DIM + kk + q * 4);
    }
  };
  auto storeStage = [&](int buf) {
    if constexpr (PRE) {
      *(uint4*)&ldsA[buf][rowP * LSTR + colP]     = qa0;
      *(uint4*)&ldsA[buf][rowP * LSTR + colP + 4] = qa1;
      *(uint4*)&ldsB[buf][rowP * LSTR + colP]     = qb0;
      *(uint4*)&ldsB[buf][rowP * LSTR + colP + 4] = qb1;
    } else {
      int row = tid >> 3, q = tid & 7;
      *(uint2*)&ldsA[buf][row * LSTR + 2 * q] =
          make_uint2(pack_bf16(pa0.x, pa0.y), pack_bf16(pa0.z, pa0.w));
      *(uint2*)&ldsB[buf][row * LSTR + 2 * q] =
          make_uint2(pack_bf16(pb0.x, pb0.y), pack_bf16(pb0.z, pb0.w));
      *(uint2*)&ldsA[buf][(row + 32) * LSTR + 2 * q] =
          make_uint2(pack_bf16(pa1.x, pa1.y), pack_bf16(pa1.z, pa1.w));
      *(uint2*)&ldsB[buf][(row + 32) * LSTR + 2 * q] =
          make_uint2(pack_bf16(pb1.x, pb1.y), pack_bf16(pb1.z, pb1.w));
      *(uint2*)&ldsA[buf][(row + 64) * LSTR + 2 * q] =
          make_uint2(pack_bf16(pa2.x, pa2.y), pack_bf16(pa2.z, pa2.w));
      *(uint2*)&ldsB[buf][(row + 64) * LSTR + 2 * q] =
          make_uint2(pack_bf16(pb2.x, pb2.y), pack_bf16(pb2.z, pb2.w));
      *(uint2*)&ldsA[buf][(row + 96) * LSTR + 2 * q] =
          make_uint2(pack_bf16(pa3.x, pa3.y), pack_bf16(pa3.z, pa3.w));
      *(uint2*)&ldsB[buf][(row + 96) * LSTR + 2 * q] =
          make_uint2(pack_bf16(pb3.x, pb3.y), pack_bf16(pb3.z, pb3.w));
    }
  };

  loadStage(0);
  storeStage(0);
  int buf = 0;
  for (int kk = 0; kk < IN_DIM; kk += 32) {
    bool hasNext = (kk + 32) < IN_DIM;
    if (hasNext) loadStage(kk + 32);
    __syncthreads();
    v16bf a[2], b[4];
#pragma unroll
    for (int i = 0; i < 2; ++i)
      a[i] = load_frag_A(ldsA[buf], waveM * 32 + i * 16 + l16, half);
#pragma unroll
    for (int j = 0; j < 4; ++j)
      b[j] = load_frag_B(ldsB[buf], waveN * 64 + j * 16 + l16, half);
#pragma unroll
    for (int i = 0; i < 2; ++i)
#pragma unroll
      for (int j = 0; j < 4; ++j)
        acc[i][j] = __builtin_amdgcn_wmma_f32_16x16x32_bf16(
            false, a[i], false, b[j], (short)0, acc[i][j], false, false);
    if (hasNext) { storeStage(buf ^ 1); buf ^= 1; }
  }

  // epilogue: bias (varies along m=h within lane) + relu -> Ht[e][b][h]
#pragma unroll
  for (int i = 0; i < 2; ++i) {
    int hBase = mBase + waveM * 32 + i * 16 + half * 8;
    float4 bb0 = *(const float4*)(fc1_b + e * HID_DIM + hBase);
    float4 bb1 = *(const float4*)(fc1_b + e * HID_DIM + hBase + 4);
    float bias[8] = {bb0.x, bb0.y, bb0.z, bb0.w, bb1.x, bb1.y, bb1.z, bb1.w};
#pragma unroll
    for (int j = 0; j < 4; ++j) {
      int bLoc = nBase + waveN * 64 + j * 16 + l16;
      Acc8 c; c.v = acc[i][j];
      unsigned pk[4];
#pragma unroll
      for (int q = 0; q < 4; ++q) {
        float v0 = c.f[2 * q] + bias[2 * q];         v0 = v0 > 0.f ? v0 : 0.f;
        float v1 = c.f[2 * q + 1] + bias[2 * q + 1]; v1 = v1 > 0.f ? v1 : 0.f;
        pk[q] = pack_bf16(v0, v1);
      }
      *(uint4*)(Ht + ((size_t)e * CHUNK + bLoc) * HID_DIM + hBase) =
          make_uint4(pk[0], pk[1], pk[2], pk[3]);
    }
  }
}

// --- GEMM2 fused: D[o][b] = W2 * Ht^T (+b2); acc[t] += gate*exp(D);
// --- out = log(acc==0 ? eps : acc). Block 64(o)x64(b), 4 waves (2Mx2N).
template <bool PRE>
__global__ __launch_bounds__(128, 2) void gemm2_combine_kernel(
    const unsigned short* __restrict__ Ht, const void* __restrict__ w2src,
    const float* __restrict__ fc2_b, const float* __restrict__ gates,
    float* __restrict__ out, int chunkBase) {
  __shared__ __attribute__((aligned(16))) unsigned ldsW[2][64 * LSTR];

  const int nBase = blockIdx.x * 64;   // batch (within chunk)
  const int mBase = blockIdx.y * 64;   // out features
  const int tid = threadIdx.x;
  const int lane = tid & 31;
  const int wave = tid >> 5;
  const int waveM = wave & 1;
  const int waveN = wave >> 1;
  const int half = lane >> 4;
  const int l16 = lane & 15;

  float tacc[TT][2][2][8];
#pragma unroll
  for (int t = 0; t < TT; ++t)
#pragma unroll
    for (int i = 0; i < 2; ++i)
#pragma unroll
      for (int j = 0; j < 2; ++j)
#pragma unroll
        for (int r = 0; r < 8; ++r) tacc[t][i][j][r] = 0.f;

  const v8f vzero = {0.f, 0.f, 0.f, 0.f, 0.f, 0.f, 0.f, 0.f};

  const unsigned short* htRow[2];
#pragma unroll
  for (int j = 0; j < 2; ++j) {
    int bLoc = nBase + waveN * 32 + j * 16 + l16;
    htRow[j] = Ht + (size_t)bLoc * HID_DIM;
  }

  const int rowP = tid >> 1;          // PRE path: 0..63
  const int colP = (tid & 1) * 8;

  for (int e = 0; e < NE; ++e) {
    const float* wF = (const float*)w2src + ((size_t)e * OUT_DIM + mBase) * HID_DIM;
    const unsigned* wU = (const unsigned*)w2src + (((size_t)e * OUT_DIM + mBase) * HID_DIM >> 1);
    const size_t eOff = (size_t)e * CHUNK * HID_DIM;

    v8f acc[2][2];
#pragma unroll
    for (int i = 0; i < 2; ++i)
#pragma unroll
      for (int j = 0; j < 2; ++j) acc[i][j] = vzero;

    float4 pw0, pw1, pw2, pw3;
    uint4 qw0, qw1;
    auto loadStage = [&](int kk) {
      if constexpr (PRE) {
        const uint4* p = (const uint4*)(wU + (size_t)rowP * (HID_DIM / 2) + (kk >> 1) + colP);
        qw0 = p[0]; qw1 = p[1];
      } else {
        int row = tid >> 3, q = tid & 7;
        pw0 = *(const float4*)(wF + (size_t)row * HID_DIM + kk + q * 4);
        pw1 = *(const float4*)(wF + (size_t)(row + 16) * HID_DIM + kk + q * 4);
        pw2 = *(const float4*)(wF + (size_t)(row + 32) * HID_DIM + kk + q * 4);
        pw3 = *(const float4*)(wF + (size_t)(row + 48) * HID_DIM + kk + q * 4);
      }
    };
    auto storeStage = [&](int buf) {
      if constexpr (PRE) {
        *(uint4*)&ldsW[buf][rowP * LSTR + colP]     = qw0;
        *(uint4*)&ldsW[buf][rowP * LSTR + colP + 4] = qw1;
      } else {
        int row = tid >> 3, q = tid & 7;
        *(uint2*)&ldsW[buf][row * LSTR + 2 * q] =
            make_uint2(pack_bf16(pw0.x, pw0.y), pack_bf16(pw0.z, pw0.w));
        *(uint2*)&ldsW[buf][(row + 16) * LSTR + 2 * q] =
            make_uint2(pack_bf16(pw1.x, pw1.y), pack_bf16(pw1.z, pw1.w));
        *(uint2*)&ldsW[buf][(row + 32) * LSTR + 2 * q] =
            make_uint2(pack_bf16(pw2.x, pw2.y), pack_bf16(pw2.z, pw2.w));
        *(uint2*)&ldsW[buf][(row + 48) * LSTR + 2 * q] =
            make_uint2(pack_bf16(pw3.x, pw3.y), pack_bf16(pw3.z, pw3.w));
      }
    };

    loadStage(0);
    storeStage(0);
    int buf = 0;
    for (int kk = 0; kk < HID_DIM; kk += 32) {
      bool hasNext = (kk + 32) < HID_DIM;
      if (hasNext) loadStage(kk + 32);
      __syncthreads();
      v16bf a[2], b[2];
#pragma unroll
      for (int i = 0; i < 2; ++i)
        a[i] = load_frag_A(ldsW[buf], waveM * 32 + i * 16 + l16, half);
#pragma unroll
      for (int j = 0; j < 2; ++j)
        b[j] = load_frag_B_global(htRow[j] + eOff, kk, half);
#pragma unroll
      for (int i = 0; i < 2; ++i)
#pragma unroll
        for (int j = 0; j < 2; ++j)
          acc[i][j] = __builtin_amdgcn_wmma_f32_16x16x32_bf16(
              false, a[i], false, b[j], (short)0, acc[i][j], false, false);
      if (hasNext) { storeStage(buf ^ 1); buf ^= 1; }
    }

    // gated exp-accumulate (gate is per-lane constant: n = batch row)
#pragma unroll
    for (int i = 0; i < 2; ++i) {
      int oBase = mBase + waveM * 32 + i * 16 + half * 8;
      float4 bb0 = *(const float4*)(fc2_b + e * OUT_DIM + oBase);
      float4 bb1 = *(const float4*)(fc2_b + e * OUT_DIM + oBase + 4);
      float bias[8] = {bb0.x, bb0.y, bb0.z, bb0.w, bb1.x, bb1.y, bb1.z, bb1.w};
#pragma unroll
      for (int j = 0; j < 2; ++j) {
        int bGlob = chunkBase + nBase + waveN * 32 + j * 16 + l16;
        const float* g = gates + ((size_t)bGlob * NE + e) * TT;
        float g0 = g[0], g1 = g[1], g2 = g[2];
        Acc8 c; c.v = acc[i][j];
#pragma unroll
        for (int r = 0; r < 8; ++r) {
          float v = __expf(c.f[r] + bias[r]);
          tacc[0][i][j][r] += g0 * v;
          tacc[1][i][j][r] += g1 * v;
          tacc[2][i][j][r] += g2 * v;
        }
      }
    }
  }

  // final: log(combined), zeros -> eps; 8 consecutive o per lane
#pragma unroll
  for (int i = 0; i < 2; ++i) {
    int oBase = mBase + waveM * 32 + i * 16 + half * 8;
#pragma unroll
    for (int j = 0; j < 2; ++j) {
      int bGlob = chunkBase + nBase + waveN * 32 + j * 16 + l16;
#pragma unroll
      for (int t = 0; t < TT; ++t) {
        float o[8];
#pragma unroll
        for (int r = 0; r < 8; ++r) {
          float v = tacc[t][i][j][r];
          v = (v == 0.f) ? EPSV : v;
          o[r] = __logf(v);
        }
        float* dst = out + ((size_t)t * BB + bGlob) * OUT_DIM + oBase;
        *(float4*)(dst)     = make_float4(o[0], o[1], o[2], o[3]);
        *(float4*)(dst + 4) = make_float4(o[4], o[5], o[6], o[7]);
      }
    }
  }
}

extern "C" void kernel_launch(void* const* d_in, const int* in_sizes, int n_in,
                              void* d_out, int out_size, void* d_ws, size_t ws_size,
                              hipStream_t stream) {
  const float* x      = (const float*)d_in[0];
  const float* w_gate = (const float*)d_in[1];
  const float* fc1_w  = (const float*)d_in[2];
  const float* fc1_b  = (const float*)d_in[3];
  const float* fc2_w  = (const float*)d_in[4];
  const float* fc2_b  = (const float*)d_in[5];
  float* out = (float*)d_out;

  char* ws = (char*)d_ws;
  const size_t logitsB = (size_t)TT * BB * NE * 4;
  const size_t gatesB  = (size_t)BB * NE * TT * 4;
  const size_t htB     = (size_t)NE * CHUNK * HID_DIM * 2;
  const size_t xbB     = (size_t)BB * IN_DIM * 2;
  const size_t w1bB    = (size_t)NE * HID_DIM * IN_DIM * 2;
  const size_t w2bB    = (size_t)NE * OUT_DIM * HID_DIM * 2;

  float* logits = (float*)ws;
  float* gates  = (float*)(ws + logitsB);
  unsigned short* Ht = (unsigned short*)(ws + logitsB + gatesB);
  char* extra = ws + logitsB + gatesB + htB;
  unsigned* Xb  = (unsigned*)extra;
  unsigned* W1b = (unsigned*)(extra + xbB);
  unsigned* W2b = (unsigned*)(extra + xbB + w1bB);
  const bool pre = ws_size >= (logitsB + gatesB + htB + xbB + w1bB + w2bB);

  gating_logits_kernel<<<(TT * BB * NE) / 256, 256, 0, stream>>>(x, w_gate, logits);
  gating_topk_kernel<<<BB / 256, 256, 0, stream>>>(logits, gates);

  if (pre) {
    cvt_bf16_kernel<<<1024, 256, 0, stream>>>(x, Xb, BB * IN_DIM / 4);
    cvt_bf16_kernel<<<4096, 256, 0, stream>>>(fc1_w, W1b, NE * HID_DIM * IN_DIM / 4);
    cvt_bf16_kernel<<<4096, 256, 0, stream>>>(fc2_w, W2b, NE * OUT_DIM * HID_DIM / 4);
  }

  for (int c = 0; c < BB / CHUNK; ++c) {
    int chunkBase = c * CHUNK;
    dim3 g1(CHUNK / 128, HID_DIM / 128, NE);
    dim3 g2(CHUNK / 64, OUT_DIM / 64);
    if (pre) {
      gemm1_kernel<true><<<g1, 256, 0, stream>>>(Xb, W1b, fc1_b, Ht, chunkBase);
      gemm2_combine_kernel<true><<<g2, 128, 0, stream>>>(Ht, W2b, fc2_b, gates, out, chunkBase);
    } else {
      gemm1_kernel<false><<<g1, 256, 0, stream>>>(x, fc1_w, fc1_b, Ht, chunkBase);
      gemm2_combine_kernel<false><<<g2, 128, 0, stream>>>(Ht, fc2_w, fc2_b, gates, out, chunkBase);
    }
  }
}
